// DetectionLoss_44083544326238
// MI455X (gfx1250) — compile-verified, matrix operand order checked by reading, not run on hardware
//
#include <hip/hip_runtime.h>
#include <hip/hip_bf16.h>

#define B_ 8
#define N_ 65536
#define C_ 80
#define T_ 32
#define MIN_POS_ 16

typedef __attribute__((ext_vector_type(16))) _Float16 v16h;
typedef __attribute__((ext_vector_type(8)))  float    v8f;

// ---------------------------------------------------------------------------
// Kernel 1: fused per-anchor pass. One wave = 16 anchors.
//   - decode + IoU vs 32 targets (16 per half-wave, combined via shfl_xor 16)
//   - sum_c exp(cls[a][c]) via v_wmma_f32_16x16x32_f16 with a ones B-matrix
//   - focal pos/neg terms + smooth-L1 sum, all selection-independent
// ---------------------------------------------------------------------------
__global__ __launch_bounds__(256)
void per_anchor_kernel(const float* __restrict__ cls,
                       const float* __restrict__ reg,
                       const float* __restrict__ anchors,
                       const float* __restrict__ tboxes,
                       const int*   __restrict__ tlabels,
                       float* __restrict__ maxiou,
                       float* __restrict__ flpos,
                       float* __restrict__ flneg,
                       float* __restrict__ sl1out)
{
    __shared__ float s_tb[T_ * 4];
    __shared__ int   s_tl[T_];
    __shared__ float s_sum[128];   // 8 waves * 16 rows

    const int blocksPerImg = N_ / 128;
    const int b    = blockIdx.x / blocksPerImg;
    const int tile = blockIdx.x % blocksPerImg;

    if (threadIdx.x < T_ * 4) s_tb[threadIdx.x] = tboxes[(size_t)b * T_ * 4 + threadIdx.x];
    if (threadIdx.x < T_)     s_tl[threadIdx.x] = tlabels[(size_t)b * T_ + threadIdx.x];
    __syncthreads();

    const int wave = threadIdx.x >> 5;
    const int lane = threadIdx.x & 31;
    const int row  = lane & 15;
    const int half = lane >> 4;
    const int a    = tile * 128 + wave * 16 + row;

    // ---- decode predicted box ----
    const float4 an = *(const float4*)(anchors + (size_t)a * 4);
    const float4 rg = *(const float4*)(reg + ((size_t)b * N_ + a) * 4);
    float aw  = an.z - an.x, ah = an.w - an.y;
    float acx = an.x + 0.5f * aw, acy = an.y + 0.5f * ah;
    float cx  = rg.x * aw + acx,  cy  = rg.y * ah + acy;
    float w   = __expf(rg.z) * aw, h  = __expf(rg.w) * ah;
    float dx1 = cx - 0.5f * w, dy1 = cy - 0.5f * h;
    float dx2 = cx + 0.5f * w, dy2 = cy + 0.5f * h;
    float areaA = (dx2 - dx1) * (dy2 - dy1);

    // ---- IoU vs 16 targets per half-wave ----
    float best = -1.0f; int bidx = 0;
    for (int t = half * 16; t < half * 16 + 16; ++t) {
        float bx1 = s_tb[t*4+0], by1 = s_tb[t*4+1], bx2 = s_tb[t*4+2], by2 = s_tb[t*4+3];
        float lx = fmaxf(dx1, bx1), ly = fmaxf(dy1, by1);
        float rx = fminf(dx2, bx2), ry = fminf(dy2, by2);
        float iw = fmaxf(rx - lx, 0.f), ih = fmaxf(ry - ly, 0.f);
        float inter = iw * ih;
        float areaB = (bx2 - bx1) * (by2 - by1);
        float iou = inter / (areaA + areaB - inter + 1e-6f);
        if (iou > best) { best = iou; bidx = t; }
    }
    {   // combine halves; tie -> lower target index (argmax first-occurrence)
        float obest = __shfl_xor(best, 16);
        int   oidx  = __shfl_xor(bidx, 16);
        if (obest > best || (obest == best && oidx < bidx)) { best = obest; bidx = oidx; }
    }

    const int label = s_tl[bidx];
    const float* cp = cls + ((size_t)b * N_ + a) * C_;
    const float xlab = cp[label];
    const float x0   = cp[0];

    // ---- smooth-L1 regression term ----
    float gx1 = s_tb[bidx*4+0], gy1 = s_tb[bidx*4+1], gx2 = s_tb[bidx*4+2], gy2 = s_tb[bidx*4+3];
    float gw = gx2 - gx1, gh = gy2 - gy1;
    float gcx = gx1 + 0.5f * gw, gcy = gy1 + 0.5f * gh;
    float t0 = (gcx - acx) / (aw + 1e-6f);
    float t1 = (gcy - acy) / (ah + 1e-6f);
    float t2 = __logf(gw / (aw + 1e-6f));
    float t3 = __logf(gh / (ah + 1e-6f));
    float sl = 0.f, d;
    d = fabsf(rg.x - t0); sl += (d < 1.f) ? 0.5f*d*d : d - 0.5f;
    d = fabsf(rg.y - t1); sl += (d < 1.f) ? 0.5f*d*d : d - 0.5f;
    d = fabsf(rg.z - t2); sl += (d < 1.f) ? 0.5f*d*d : d - 0.5f;
    d = fabsf(rg.w - t3); sl += (d < 1.f) ? 0.5f*d*d : d - 0.5f;

    // ---- sum_c exp(x) via WMMA: A = exp(logits) as f16, B = ones ----
    v8f acc = {};
    v16h ones;
    #pragma unroll
    for (int i = 0; i < 16; ++i) ones[i] = (_Float16)1.0f;

    #pragma unroll
    for (int c = 0; c < 3; ++c) {
        const int g0 = c * 32 + half * 8;        // K 0..7 of this half
        const int g1 = c * 32 + 16 + half * 8;   // K 16..23 of this half
        v16h A;
        float4 u0 = *(const float4*)(cp + g0);
        float4 u1 = *(const float4*)(cp + g0 + 4);
        A[0] = (_Float16)__expf(u0.x); A[1] = (_Float16)__expf(u0.y);
        A[2] = (_Float16)__expf(u0.z); A[3] = (_Float16)__expf(u0.w);
        A[4] = (_Float16)__expf(u1.x); A[5] = (_Float16)__expf(u1.y);
        A[6] = (_Float16)__expf(u1.z); A[7] = (_Float16)__expf(u1.w);
        if (c < 2) {   // g1 < 80 only for c = 0,1 (uniform per chunk)
            float4 u2 = *(const float4*)(cp + g1);
            float4 u3 = *(const float4*)(cp + g1 + 4);
            A[8]  = (_Float16)__expf(u2.x); A[9]  = (_Float16)__expf(u2.y);
            A[10] = (_Float16)__expf(u2.z); A[11] = (_Float16)__expf(u2.w);
            A[12] = (_Float16)__expf(u3.x); A[13] = (_Float16)__expf(u3.y);
            A[14] = (_Float16)__expf(u3.z); A[15] = (_Float16)__expf(u3.w);
        } else {       // pad K 80..95 with zeros
            #pragma unroll
            for (int j = 8; j < 16; ++j) A[j] = (_Float16)0.0f;
        }
        acc = __builtin_amdgcn_wmma_f32_16x16x32_f16(
                  false, A, false, ones, (short)0, acc, false, false);
    }

    // D[i][j] identical for all j; row i lives in VGPR i%8 of lane 0 / lane 16
    if (lane == 0) {
        #pragma unroll
        for (int r = 0; r < 8; ++r) s_sum[wave * 16 + r] = acc[r];
    }
    if (lane == 16) {
        #pragma unroll
        for (int r = 0; r < 8; ++r) s_sum[wave * 16 + 8 + r] = acc[r];
    }
    __syncthreads();

    float sumexp = s_sum[wave * 16 + row];
    float lse    = __logf(sumexp);
    float ce_p   = lse - xlab;
    float ce_n   = lse - x0;
    float ptp = __expf(-ce_p), om = 1.f - ptp;
    float fl_p = 0.25f * om * om * ce_p;            // alpha=0.25, gamma=2, labels>0 always
    float ptn = __expf(-ce_n), on = 1.f - ptn;
    float fl_n = 0.9f * on * on * on * ce_n;        // alpha=0.1 -> 1-alpha, gamma=3

    if (half == 0) {
        size_t o = (size_t)b * N_ + a;
        maxiou[o] = best;
        flpos[o]  = fl_p;
        flneg[o]  = fl_n;
        sl1out[o] = sl;
    }
}

// ---------------------------------------------------------------------------
// Kernel 2 helpers: exact top-k via byte radix select + stable tie-break scan
// ---------------------------------------------------------------------------
__device__ unsigned radix_select(const float* __restrict__ key,
                                 const unsigned char* __restrict__ m,
                                 int n, int k, bool eligAll,
                                 int* s_hist, int* s_bcast, int* needEqOut)
{
    unsigned prefix = 0;
    int kk = k;
    for (int p = 3; p >= 0; --p) {
        for (int i = threadIdx.x; i < 256; i += blockDim.x) s_hist[i] = 0;
        __syncthreads();
        const unsigned shiftHi = (unsigned)(p + 1) * 8u;
        for (int i = threadIdx.x; i < n; i += blockDim.x) {
            if (!eligAll && m[i] != 0) continue;
            unsigned u = __float_as_uint(key[i]);
            if (p < 3 && (u >> shiftHi) != prefix) continue;
            atomicAdd(&s_hist[(u >> (p * 8)) & 255], 1);
        }
        __syncthreads();
        if (threadIdx.x == 0) {
            int c = 0, v;
            for (v = 255; v > 0; --v) {
                if (c + s_hist[v] >= kk) break;
                c += s_hist[v];
            }
            s_bcast[0] = v;
            s_bcast[1] = kk - c;
        }
        __syncthreads();
        prefix = (prefix << 8) | (unsigned)s_bcast[0];
        kk = s_bcast[1];
        __syncthreads();
    }
    *needEqOut = kk;
    return prefix;
}

__device__ void stable_mark(const float* __restrict__ key,
                            unsigned char* __restrict__ m,
                            int n, unsigned thr, int needEq,
                            bool eligMaskZero, unsigned char markVal,
                            int* s_scan)
{
    const int per = n / blockDim.x;             // contiguous per-thread range
    const int t  = threadIdx.x;
    const int lo = t * per, hi = lo + per;
    int cnt = 0;
    for (int i = lo; i < hi; ++i) {
        bool elig = eligMaskZero ? (m[i] == 0) : true;
        if (elig && __float_as_uint(key[i]) == thr) cnt++;
    }
    s_scan[t] = cnt;
    __syncthreads();
    for (int off = 1; off < (int)blockDim.x; off <<= 1) {   // inclusive scan
        int v = (t >= off) ? s_scan[t - off] : 0;
        __syncthreads();
        s_scan[t] += v;
        __syncthreads();
    }
    int base = s_scan[t] - cnt;                 // exclusive prefix
    __syncthreads();
    int c = 0;
    for (int i = lo; i < hi; ++i) {
        bool elig = eligMaskZero ? (m[i] == 0) : true;
        if (elig && __float_as_uint(key[i]) == thr) {
            if (base + c < needEq) m[i] = markVal;
            c++;
        }
    }
    __syncthreads();
}

// ---------------------------------------------------------------------------
// Kernel 2: one block per image — positive selection (with top-16 fallback),
// hard-negative mining (top 4*num_pos), masked loss reductions.
// ---------------------------------------------------------------------------
__global__ __launch_bounds__(1024)
void select_reduce_kernel(const float* __restrict__ maxiou,
                          const float* __restrict__ flpos,
                          const float* __restrict__ flneg,
                          const float* __restrict__ sl1,
                          unsigned char* __restrict__ mask,
                          float* __restrict__ perimg)   // [3*B]: cls_l, reg_l, npos
{
    __shared__ int   s_hist[256];
    __shared__ int   s_scan[1024];
    __shared__ int   s_i[4];
    __shared__ float s_f[4];

    const int b = blockIdx.x;
    const float* key = maxiou + (size_t)b * N_;
    unsigned char* m = mask + (size_t)b * N_;

    // ---- count anchors above positive IoU threshold ----
    int cnt = 0;
    for (int i = threadIdx.x; i < N_; i += blockDim.x)
        if (key[i] >= 0.5f) cnt++;
    if (threadIdx.x == 0) s_i[0] = 0;
    __syncthreads();
    atomicAdd(&s_i[0], cnt);
    __syncthreads();

    int numPos;
    if (s_i[0] >= MIN_POS_) {                    // uniform branch (shared value)
        numPos = s_i[0];
        for (int i = threadIdx.x; i < N_; i += blockDim.x)
            m[i] = (key[i] >= 0.5f) ? (unsigned char)1 : (unsigned char)0;
        __syncthreads();
    } else {                                     // top-16 fallback
        numPos = MIN_POS_;
        int needEq;
        unsigned thr = radix_select(key, m, N_, MIN_POS_, true, s_hist, &s_i[1], &needEq);
        for (int i = threadIdx.x; i < N_; i += blockDim.x)
            m[i] = (__float_as_uint(key[i]) > thr) ? (unsigned char)1 : (unsigned char)0;
        __syncthreads();
        stable_mark(key, m, N_, thr, needEq, false, 1, s_scan);
    }

    // ---- hard negatives: top k = min(N-numPos, 4*numPos) among non-pos ----
    int kneg = 4 * numPos;
    if (kneg > N_ - numPos) kneg = N_ - numPos;
    if (kneg > 0) {
        int needEq;
        unsigned thr = radix_select(key, m, N_, kneg, false, s_hist, &s_i[1], &needEq);
        for (int i = threadIdx.x; i < N_; i += blockDim.x)
            if (m[i] == 0 && __float_as_uint(key[i]) > thr) m[i] = 2;
        __syncthreads();
        stable_mark(key, m, N_, thr, needEq, true, 2, s_scan);
    }

    // ---- masked reductions ----
    float cP = 0.f, cN = 0.f, rS = 0.f;
    for (int i = threadIdx.x; i < N_; i += blockDim.x) {
        unsigned char mm = m[i];
        size_t o = (size_t)b * N_ + i;
        if (mm == 1)      { cP += flpos[o]; rS += sl1[o]; }
        else if (mm == 2) { cN += flneg[o]; }
    }
    if (threadIdx.x == 0) { s_f[0] = 0.f; s_f[1] = 0.f; s_f[2] = 0.f; }
    __syncthreads();
    atomicAdd(&s_f[0], cP);
    atomicAdd(&s_f[1], cN);
    atomicAdd(&s_f[2], rS);
    __syncthreads();
    if (threadIdx.x == 0) {
        int total = numPos + kneg; if (total < 1) total = 1;
        perimg[b]          = (s_f[0] + s_f[1]) / (float)total;
        perimg[B_ + b]     = s_f[2] / ((float)numPos + 1e-6f);
        perimg[2 * B_ + b] = (float)numPos;
    }
}

// ---------------------------------------------------------------------------
// Kernel 3: final scalar reduction -> (total, cls_final, reg_final, total_pos)
// ---------------------------------------------------------------------------
__global__ void finalize_kernel(const float* __restrict__ perimg,
                                float* __restrict__ out)
{
    if (threadIdx.x == 0 && blockIdx.x == 0) {
        float cs = 0.f, rs = 0.f, np = 0.f;
        for (int b = 0; b < B_; ++b) {
            cs += perimg[b];
            rs += perimg[B_ + b];
            np += perimg[2 * B_ + b];
        }
        float cls_final = cs / (float)B_;
        float reg_final = (np > 0.f) ? rs / (float)B_ : 0.f;
        float reg_w = fminf(1.0f, np / (100.0f * (float)B_));
        out[0] = cls_final + reg_w * 1.0f /*REG_LOSS_WEIGHT*/ * reg_final;
        out[1] = cls_final;
        out[2] = reg_final;
        out[3] = np;
    }
}

// ---------------------------------------------------------------------------
extern "C" void kernel_launch(void* const* d_in, const int* in_sizes, int n_in,
                              void* d_out, int out_size, void* d_ws, size_t ws_size,
                              hipStream_t stream) {
    const float* cls      = (const float*)d_in[0];   // [B,N,C]
    const float* reg      = (const float*)d_in[1];   // [B,N,4]
    const float* anchors  = (const float*)d_in[2];   // [N,4]
    const float* tboxes   = (const float*)d_in[3];   // [B,T,4]
    const int*   tlabels  = (const int*)d_in[4];     // [B,T]
    float* out = (float*)d_out;

    // Workspace layout
    float* maxiou = (float*)d_ws;                    // B*N
    float* flpos  = maxiou + (size_t)B_ * N_;        // B*N
    float* flneg  = flpos  + (size_t)B_ * N_;        // B*N
    float* sl1    = flneg  + (size_t)B_ * N_;        // B*N
    float* perimg = sl1    + (size_t)B_ * N_;        // 3*B
    unsigned char* mask = (unsigned char*)(perimg + 3 * B_);  // B*N bytes

    per_anchor_kernel<<<B_ * (N_ / 128), 256, 0, stream>>>(
        cls, reg, anchors, tboxes, tlabels, maxiou, flpos, flneg, sl1);

    select_reduce_kernel<<<B_, 1024, 0, stream>>>(
        maxiou, flpos, flneg, sl1, mask, perimg);

    finalize_kernel<<<1, 32, 0, stream>>>(perimg, out);
}
